// MolecularGraphEncoder_48747878810102
// MI455X (gfx1250) — compile-verified, matrix operand order checked by reading, not run on hardware
//
#include <hip/hip_runtime.h>
#include <hip/hip_bf16.h>
#include <stdint.h>

#define N_NODES 100000
#define N_EDGES 1600000
#define F_IN    78
#define F_PAD   96        // F padded to multiple of 32 for WMMA K-steps
#define H_DIM   256
#define G_BATCH 2048
#define BN_EPS  1e-5f

typedef __attribute__((ext_vector_type(16))) __bf16 v16bf;
typedef __attribute__((ext_vector_type(8)))  float  v8f;

// ---- float -> bf16, round-to-nearest-even (bit math; used only in prep kernels) ----
static __device__ __forceinline__ __bf16 f2bf(float f) {
    unsigned u = __builtin_bit_cast(unsigned, f);
    u += 0x7FFFu + ((u >> 16) & 1u);
    unsigned short s = (unsigned short)(u >> 16);
    return __builtin_bit_cast(__bf16, s);
}
static __device__ __forceinline__ __bf16 bf_zero() {
    return __builtin_bit_cast(__bf16, (unsigned short)0);
}

// ---- 16-element bf16 fragment = two aligned 16-byte loads ----
struct frag32 { uint4 lo; uint4 hi; };
static __device__ __forceinline__ v16bf load_frag(const __bf16* p0, const __bf16* p1) {
    frag32 f;
    f.lo = *(const uint4*)p0;
    f.hi = *(const uint4*)p1;
    return __builtin_bit_cast(v16bf, f);
}

// =====================  degree / norm  =====================
__global__ void k_deg_init(float* deg, int n) {
    int i = blockIdx.x * 256 + threadIdx.x;
    if (i < n) deg[i] = 1.0f;
}
__global__ void k_deg_count(const long long* __restrict__ dst, float* deg, int e) {
    int i = blockIdx.x * 256 + threadIdx.x;
    if (i < e) atomicAdd(&deg[dst[i]], 1.0f);
}
__global__ void k_deg_rsqrt(float* deg, int n) {
    int i = blockIdx.x * 256 + threadIdx.x;
    if (i < n) deg[i] = rsqrtf(deg[i]);
}

// =====================  operand packing  =====================
// x [N][78] f32 -> xp [N][96] bf16 zero-padded.  blockDim.x == 96
__global__ void k_pack_x(const float* __restrict__ x, __bf16* __restrict__ xp, int n) {
    int c = threadIdx.x;
    for (int r = blockIdx.x; r < n; r += gridDim.x)
        xp[(size_t)r * F_PAD + c] = (c < F_IN) ? f2bf(x[(size_t)r * F_IN + c]) : bf_zero();
}
// W [K][256] f32 -> Wtb [256][Kp] bf16 (transposed, zero-padded along K)
__global__ void k_pack_wt(const float* __restrict__ W, __bf16* __restrict__ Wtb,
                          int K, int Kp) {
    int idx = blockIdx.x * 256 + threadIdx.x;
    if (idx < H_DIM * Kp) {
        int nn = idx / Kp, k = idx % Kp;
        Wtb[idx] = (k < K) ? f2bf(W[(size_t)k * H_DIM + nn]) : bf_zero();
    }
}

// =====================  WMMA GEMM: C[M,256] = A[M,Kp] * W[Kp,256]  =====================
// A: bf16 [M][Kp] (lda==Kp), Bt: bf16 [256][Kp] (W transposed), C: f32 [M][256].
// One wave computes a 16x64 strip: 1 A fragment x 4 B fragments per k-step.
__global__ void k_gemm_wmma(const __bf16* __restrict__ A, int Kp,
                            const __bf16* __restrict__ Bt,
                            float* __restrict__ C, int mtiles) {
    int wave = blockIdx.x * (blockDim.x >> 5) + (threadIdx.x >> 5);
    int lane = threadIdx.x & 31;
    if (wave >= mtiles * 4) return;           // uniform per-wave (EXEC stays all-ones)
    int mt = wave >> 2, ng = wave & 3;
    int m0 = mt * 16, n0 = ng * 64;
    int rsel = lane & 15;
    int hi   = lane >> 4;
    const __bf16* arow = A  + (size_t)(m0 + rsel) * Kp;
    const __bf16* b0   = Bt + (size_t)(n0 + rsel) * Kp;
    const __bf16* b1   = b0 + (size_t)16 * Kp;
    const __bf16* b2   = b0 + (size_t)32 * Kp;
    const __bf16* b3   = b0 + (size_t)48 * Kp;
    v8f c0 = {}, c1 = {}, c2 = {}, c3 = {};
    for (int kc = 0; kc < Kp; kc += 32) {
        int kb0 = kc + hi * 8;
        int kb1 = kc + 16 + hi * 8;
        v16bf a  = load_frag(arow + kb0, arow + kb1);
        v16bf f0 = load_frag(b0 + kb0, b0 + kb1);
        v16bf f1 = load_frag(b1 + kb0, b1 + kb1);
        v16bf f2 = load_frag(b2 + kb0, b2 + kb1);
        v16bf f3 = load_frag(b3 + kb0, b3 + kb1);
        c0 = __builtin_amdgcn_wmma_f32_16x16x32_bf16(false, a, false, f0, (short)0, c0, false, false);
        c1 = __builtin_amdgcn_wmma_f32_16x16x32_bf16(false, a, false, f1, (short)0, c1, false, false);
        c2 = __builtin_amdgcn_wmma_f32_16x16x32_bf16(false, a, false, f2, (short)0, c2, false, false);
        c3 = __builtin_amdgcn_wmma_f32_16x16x32_bf16(false, a, false, f3, (short)0, c3, false, false);
    }
    // C/D layout: VGPR r <-> row (r + 8*hi), col = rsel
    float* crow = C + (size_t)(m0 + hi * 8) * H_DIM + n0 + rsel;
#pragma unroll
    for (int r = 0; r < 8; ++r) {
        crow[(size_t)r * H_DIM +  0] = c0[r];
        crow[(size_t)r * H_DIM + 16] = c1[r];
        crow[(size_t)r * H_DIM + 32] = c2[r];
        crow[(size_t)r * H_DIM + 48] = c3[r];
    }
}

// =====================  aggregation  =====================
__global__ void k_agg_init(const float* __restrict__ hw, const float* __restrict__ dis,
                           const float* __restrict__ bias, float* __restrict__ agg, int n) {
    int c = threadIdx.x;
    float bc = bias[c];
    for (int r = blockIdx.x; r < n; r += gridDim.x) {
        float sn = dis[r]; sn *= sn;
        agg[(size_t)r * H_DIM + c] = hw[(size_t)r * H_DIM + c] * sn + bc;
    }
}
__global__ void k_agg_edges(const long long* __restrict__ src, const long long* __restrict__ dst,
                            const float* __restrict__ dis, const float* __restrict__ hw,
                            float* __restrict__ agg) {
    int e = blockIdx.x;
    int c = threadIdx.x;
    long long s = src[e], d = dst[e];
    float w = dis[s] * dis[d];
    atomicAdd(&agg[(size_t)d * H_DIM + c], hw[(size_t)s * H_DIM + c] * w);
}

// =====================  batchnorm (+ReLU, + bf16 repack for next GEMM)  =====================
__global__ void k_zero(float* p, int n) {
    int i = blockIdx.x * 256 + threadIdx.x;
    if (i < n) p[i] = 0.0f;
}
__global__ void k_bn_stats(const float* __restrict__ h, float* __restrict__ sums, int n) {
    int c = threadIdx.x;
    int r0 = blockIdx.x * 64;
    int r1 = r0 + 64; if (r1 > n) r1 = n;
    float s = 0.0f, q = 0.0f;
    for (int r = r0; r < r1; ++r) {
        float v = h[(size_t)r * H_DIM + c];
        s += v; q += v * v;
    }
    atomicAdd(&sums[c], s);
    atomicAdd(&sums[H_DIM + c], q);
}
__global__ void k_bn_apply(float* __restrict__ h, __bf16* __restrict__ hb,
                           const float* __restrict__ sums,
                           const float* __restrict__ g, const float* __restrict__ be, int n) {
    int c = threadIdx.x;
    float invn = 1.0f / (float)n;
    float mu  = sums[c] * invn;
    float var = sums[H_DIM + c] * invn - mu * mu;
    float scl = rsqrtf(var + BN_EPS) * g[c];
    float sh  = be[c] - mu * scl;
    for (int r = blockIdx.x; r < n; r += gridDim.x) {
        size_t idx = (size_t)r * H_DIM + c;
        float v = h[idx] * scl + sh;
        v = v > 0.0f ? v : 0.0f;
        h[idx] = v;
        hb[idx] = f2bf(v);
    }
}

// =====================  global mean pool  =====================
__global__ void k_counts(const long long* __restrict__ batch, float* counts, int n) {
    int i = blockIdx.x * 256 + threadIdx.x;
    if (i < n) atomicAdd(&counts[batch[i]], 1.0f);
}
__global__ void k_pool_sum(const float* __restrict__ h, const long long* __restrict__ batch,
                           float* __restrict__ out, int n) {
    int c = threadIdx.x;
    for (int r = blockIdx.x; r < n; r += gridDim.x)
        atomicAdd(&out[(size_t)batch[r] * H_DIM + c], h[(size_t)r * H_DIM + c]);
}
__global__ void k_pool_div(float* __restrict__ out, const float* __restrict__ counts) {
    int g = blockIdx.x, c = threadIdx.x;
    float cnt = counts[g]; if (cnt < 1.0f) cnt = 1.0f;
    out[(size_t)g * H_DIM + c] /= cnt;
}

// =====================  driver  =====================
extern "C" void kernel_launch(void* const* d_in, const int* in_sizes, int n_in,
                              void* d_out, int out_size, void* d_ws, size_t ws_size,
                              hipStream_t stream) {
    const float*     x  = (const float*)d_in[0];
    const long long* ei = (const long long*)d_in[1];
    const long long* bt = (const long long*)d_in[2];
    const float* W1 = (const float*)d_in[3];  const float* b1 = (const float*)d_in[4];
    const float* W2 = (const float*)d_in[5];  const float* b2 = (const float*)d_in[6];
    const float* W3 = (const float*)d_in[7];  const float* b3 = (const float*)d_in[8];
    const float* g1 = (const float*)d_in[9];  const float* be1 = (const float*)d_in[10];
    const float* g2 = (const float*)d_in[11]; const float* be2 = (const float*)d_in[12];
    const float* g3 = (const float*)d_in[13]; const float* be3 = (const float*)d_in[14];
    float* out = (float*)d_out;

    const int N = N_NODES, E = N_EDGES, H = H_DIM, G = G_BATCH;
    const long long* src = ei;
    const long long* dst = ei + E;

    // workspace layout (all offsets 256B-aligned)
    char* p = (char*)d_ws;
    auto bump = [&](size_t bytes) { char* q = p; p += (bytes + 255) & ~size_t(255); return q; };
    float*  dis    = (float*) bump((size_t)N * 4);
    float*  bufA   = (float*) bump((size_t)N * H * 4);   // agg / activations f32
    float*  bufB   = (float*) bump((size_t)N * H * 4);   // h @ W f32
    __bf16* bufC   = (__bf16*)bump((size_t)N * H * 2);   // bf16 GEMM input (aliases x-pack)
    __bf16* Wtb    = (__bf16*)bump((size_t)H * H * 2);   // transposed bf16 weights
    float*  stats  = (float*) bump(2 * H * 4);
    float*  counts = (float*) bump((size_t)G * 4);
    (void)ws_size; (void)n_in; (void)in_sizes; (void)out_size;

    const int mtiles = N / 16;                        // 6250 (exact)
    const int gemmBlocks = mtiles;                    // 4 waves/block, 4 waves per m-tile
    const int gsRows = 4096;

    // degrees with self-loops -> deg^{-1/2}
    k_deg_init <<<(N + 255) / 256, 256, 0, stream>>>(dis, N);
    k_deg_count<<<(E + 255) / 256, 256, 0, stream>>>(dst, dis, E);
    k_deg_rsqrt<<<(N + 255) / 256, 256, 0, stream>>>(dis, N);

    // ---------------- layer 1 ----------------
    k_pack_x   <<<gsRows, F_PAD, 0, stream>>>(x, bufC, N);
    k_pack_wt  <<<(H * F_PAD + 255) / 256, 256, 0, stream>>>(W1, Wtb, F_IN, F_PAD);
    k_gemm_wmma<<<gemmBlocks, 128, 0, stream>>>(bufC, F_PAD, Wtb, bufB, mtiles);
    k_agg_init <<<gsRows, H, 0, stream>>>(bufB, dis, b1, bufA, N);
    k_agg_edges<<<E, H, 0, stream>>>(src, dst, dis, bufB, bufA);
    k_zero     <<<(2 * H + 255) / 256, 256, 0, stream>>>(stats, 2 * H);
    k_bn_stats <<<(N + 63) / 64, H, 0, stream>>>(bufA, stats, N);
    k_bn_apply <<<gsRows, H, 0, stream>>>(bufA, bufC, stats, g1, be1, N);

    // ---------------- layer 2 ----------------
    k_pack_wt  <<<(H * H + 255) / 256, 256, 0, stream>>>(W2, Wtb, H, H);
    k_gemm_wmma<<<gemmBlocks, 128, 0, stream>>>(bufC, H, Wtb, bufB, mtiles);
    k_agg_init <<<gsRows, H, 0, stream>>>(bufB, dis, b2, bufA, N);
    k_agg_edges<<<E, H, 0, stream>>>(src, dst, dis, bufB, bufA);
    k_zero     <<<(2 * H + 255) / 256, 256, 0, stream>>>(stats, 2 * H);
    k_bn_stats <<<(N + 63) / 64, H, 0, stream>>>(bufA, stats, N);
    k_bn_apply <<<gsRows, H, 0, stream>>>(bufA, bufC, stats, g2, be2, N);

    // ---------------- layer 3 ----------------
    k_pack_wt  <<<(H * H + 255) / 256, 256, 0, stream>>>(W3, Wtb, H, H);
    k_gemm_wmma<<<gemmBlocks, 128, 0, stream>>>(bufC, H, Wtb, bufB, mtiles);
    k_agg_init <<<gsRows, H, 0, stream>>>(bufB, dis, b3, bufA, N);
    k_agg_edges<<<E, H, 0, stream>>>(src, dst, dis, bufB, bufA);
    k_zero     <<<(2 * H + 255) / 256, 256, 0, stream>>>(stats, 2 * H);
    k_bn_stats <<<(N + 63) / 64, H, 0, stream>>>(bufA, stats, N);
    k_bn_apply <<<gsRows, H, 0, stream>>>(bufA, bufC, stats, g3, be3, N);

    // ---------------- global mean pool ----------------
    k_zero     <<<(G * H + 255) / 256, 256, 0, stream>>>(out, G * H);
    k_zero     <<<(G + 255) / 256, 256, 0, stream>>>(counts, G);
    k_counts   <<<(N + 255) / 256, 256, 0, stream>>>(bt, counts, N);
    k_pool_sum <<<gsRows, H, 0, stream>>>(bufA, bt, out, N);
    k_pool_div <<<G, H, 0, stream>>>(out, counts);
}
